// RSSM_32332513804937
// MI455X (gfx1250) — matrix operand order repaired and checked by reading, not compiled, not run on hardware
//
#include <hip/hip_runtime.h>

// ---------------------------------------------------------------------------
// RSSM rollout for MI455X (gfx1250, wave32, WMMA).
// One workgroup (256 thr = 8 waves) owns 16 batch rows for all 64 timesteps;
// the scan is row-independent so no grid sync is needed. deter carry (f32) and
// activations live in LDS; all GEMMs are v_wmma_f32_16x16x32_bf16 (f32 accum).
// Weights (~11MB f32, L2-resident) are pre-converted once per launch to bf16
// in d_ws (prep kernel) when ws_size allows; otherwise converted on the fly.
// ---------------------------------------------------------------------------

typedef __attribute__((ext_vector_type(16))) __bf16 v16bf;
typedef __attribute__((ext_vector_type(8)))  float  v8f;

#define B_TOT   256
#define T_TOT   64
#define STOCH   32
#define DETER   512
#define HIDDEN  512
#define EMBED   1024
#define ACT_DIM 6

// LDS row strides padded so row offset % 64 banks == 4 (conflict-free gather)
#define DSTR 516   // f32 deter rows
#define HSTR 520   // bf16 hidden rows
#define XSTR 72    // bf16 x_in rows (K padded 38 -> 64)

// bf16 weight workspace layout (elements)
constexpr size_t S1P = 64 * 512;        // W_img1 zero-padded K 38->64
constexpr size_t SIH = 512 * 1536;
constexpr size_t SHH = 512 * 1536;
constexpr size_t S2  = 512 * 512;
constexpr size_t S3  = 512 * 64;
constexpr size_t SO1 = 1536 * 512;
constexpr size_t SO2 = 512 * 64;
constexpr size_t WTOT = S1P + SIH + SHH + S2 + S3 + SO1 + SO2;

// -------- fast activations (v_exp_f32 / v_log_f32 TRANS ops, co-exec w/ WMMA)
__device__ __forceinline__ float sigm(float x)  { return 1.0f / (1.0f + __expf(-x)); }
__device__ __forceinline__ float eluf(float x)  { return x > 0.0f ? x : __expf(x) - 1.0f; }
__device__ __forceinline__ float softpf(float x){ return x > 20.0f ? x : __logf(1.0f + __expf(x)); }
__device__ __forceinline__ float tanh_f(float x){
    float e = __expf(-2.0f * fabsf(x));
    float t = (1.0f - e) / (1.0f + e);
    return copysignf(t, x);
}

__device__ __forceinline__ v8f wmma_bf(v16bf a, v16bf b, v8f c) {
    // D = A(16x32 bf16) * B(32x16 bf16) + C(16x16 f32)
    return __builtin_amdgcn_wmma_f32_16x16x32_bf16(false, a, false, b,
                                                   (short)0, c, false, false);
}

// A fragment from bf16 row-major LDS: lane holds row M=ln, K per ISA layout:
// elem j -> K = (j>>3)*16 + hl*8 + (j&7)
__device__ __forceinline__ v16bf load_a_bf16(const __bf16* __restrict__ s,
                                             int stride, int ln, int hl) {
    const __bf16* p = s + ln * stride + hl * 8;
    v16bf a;
#pragma unroll
    for (int j = 0; j < 8; ++j) a[j] = p[j];
#pragma unroll
    for (int j = 0; j < 8; ++j) a[8 + j] = p[16 + j];
    return a;
}

// A fragment from f32 row-major (LDS or global), converted to bf16
__device__ __forceinline__ v16bf load_a_f32(const float* __restrict__ s,
                                            int stride, int ln, int hl) {
    const float* p = s + (size_t)ln * stride + hl * 8;
    v16bf a;
#pragma unroll
    for (int j = 0; j < 8; ++j) a[j] = (__bf16)p[j];
#pragma unroll
    for (int j = 0; j < 8; ++j) a[8 + j] = (__bf16)p[16 + j];
    return a;
}

// B fragment: lane holds column N=lane&15, elem j -> K = (j>>3)*16 + (lane>>4)*8 + (j&7) (row-major W)
__device__ __forceinline__ v16bf load_b(const float* __restrict__ w,
                                        int ldw, int lane) {
    const float* p = w + (lane & 15) + (size_t)((lane >> 4) * 8) * ldw;
    v16bf b;
#pragma unroll
    for (int j = 0; j < 16; ++j) b[j] = (__bf16)p[(size_t)((j >> 3) * 16 + (j & 7)) * ldw];
    return b;
}
__device__ __forceinline__ v16bf load_b(const __bf16* __restrict__ w,
                                        int ldw, int lane) {
    const __bf16* p = w + (lane & 15) + (size_t)((lane >> 4) * 8) * ldw;
    v16bf b;
#pragma unroll
    for (int j = 0; j < 16; ++j) b[j] = p[(size_t)((j >> 3) * 16 + (j & 7)) * ldw];
    return b;
}

// W_img1 B fragment: f32 path needs K zero-padding (K=38), bf16 path is padded
__device__ __forceinline__ v16bf load_b_w1(const float* __restrict__ W1,
                                           int kb, int n0, int lane) {
    int krem = 38 - kb; if (krem > 32) krem = 32;
    v16bf b;
#pragma unroll
    for (int j = 0; j < 16; ++j) {
        int kk = (j >> 3) * 16 + (lane >> 4) * 8 + (j & 7);
        b[j] = (kk < krem) ? (__bf16)W1[(size_t)(kb + kk) * HIDDEN + n0 + (lane & 15)] : (__bf16)0.0f;
    }
    return b;
}
__device__ __forceinline__ v16bf load_b_w1(const __bf16* __restrict__ W1p,
                                           int kb, int n0, int lane) {
    return load_b(W1p + (size_t)kb * HIDDEN + n0, HIDDEN, lane);
}

// -------- prep: f32 weights -> bf16 workspace (W_img1 zero-padded to K=64) --
__global__ __launch_bounds__(256)
void prep_weights_kernel(const float* __restrict__ W1,  const float* __restrict__ Wih,
                         const float* __restrict__ Whh, const float* __restrict__ W2,
                         const float* __restrict__ W3,  const float* __restrict__ Wo1,
                         const float* __restrict__ Wo2, __bf16* __restrict__ ws) {
    for (size_t i = (size_t)blockIdx.x * blockDim.x + threadIdx.x; i < WTOT;
         i += (size_t)gridDim.x * blockDim.x) {
        size_t j = i;
        float v;
        if (j < S1P) {
            size_t k = j >> 9, n = j & 511;
            v = (k < 38) ? W1[k * 512 + n] : 0.0f;
        } else if ((j -= S1P) < SIH) { v = Wih[j];
        } else if ((j -= SIH) < SHH) { v = Whh[j];
        } else if ((j -= SHH) < S2)  { v = W2[j];
        } else if ((j -= S2)  < S3)  { v = W3[j];
        } else if ((j -= S3)  < SO1) { v = Wo1[j];
        } else                       { v = Wo2[j - SO1]; }
        ws[i] = (__bf16)v;
    }
}

// -------- main rollout -------------------------------------------------------
template <typename WT>
__global__ __launch_bounds__(256)
void rssm_rollout_kernel(const float* __restrict__ embed,
                         const float* __restrict__ action,
                         const WT* __restrict__ W1,  const float* __restrict__ b1,
                         const WT* __restrict__ Wih, const float* __restrict__ bih,
                         const WT* __restrict__ Whh, const float* __restrict__ bhh,
                         const WT* __restrict__ W2,  const float* __restrict__ b2,
                         const WT* __restrict__ W3,  const float* __restrict__ b3,
                         const WT* __restrict__ Wo1, const float* __restrict__ bo1,
                         const WT* __restrict__ Wo2, const float* __restrict__ bo2,
                         const float* __restrict__ npr, const float* __restrict__ npo,
                         float* __restrict__ out) {
    __shared__ float  sdet[16 * DSTR];   // deter carry (f32)
    __shared__ __bf16 shid[16 * HSTR];   // h1 / h2 / h3 (sequentially reused)
    __shared__ __bf16 sxin[16 * XSTR];   // [stoch | action | zero-pad] (K=64)

    const int tid  = threadIdx.x;
    const int wave = tid >> 5;
    const int lane = tid & 31;
    const int hl   = lane >> 4;   // lane half (0/1)
    const int ln   = lane & 15;
    const int b0   = blockIdx.x * 16;

    for (int i = tid; i < 16 * DSTR; i += 256) sdet[i] = 0.0f;
    for (int i = tid; i < 16 * XSTR; i += 256) sxin[i] = (__bf16)0.0f;
    __syncthreads();

    const int OUT32 = B_TOT * T_TOT * STOCH;
    float* outOM = out;
    float* outOS = out + 1 * OUT32;
    float* outOX = out + 2 * OUT32;
    float* outPM = out + 3 * OUT32;
    float* outPS = out + 4 * OUT32;
    float* outPX = out + 5 * OUT32;
    float* outDT = out + 6 * OUT32;

    for (int t = 0; t < T_TOT; ++t) {
        // ---- P0: stage action into x_in cols 32..37 --------------------
        if (tid < 16 * ACT_DIM) {
            int m = tid / ACT_DIM, j = tid - m * ACT_DIM;
            sxin[m * XSTR + STOCH + j] =
                (__bf16)action[((size_t)(b0 + m) * T_TOT + t) * ACT_DIM + j];
        }
        __syncthreads();

        // ---- P1: h1 = elu(x_in @ W_img1 + b_img1)   [16x64 x 64x512] ---
        {
            v16bf a0 = load_a_bf16(sxin,      XSTR, ln, hl);   // invariant over tiles
            v16bf a1 = load_a_bf16(sxin + 32, XSTR, ln, hl);
#pragma unroll
            for (int i = 0; i < 4; ++i) {
                int n0 = (wave * 4 + i) * 16;
                v8f acc = {};
                acc = wmma_bf(a0, load_b_w1(W1, 0,  n0, lane), acc);
                acc = wmma_bf(a1, load_b_w1(W1, 32, n0, lane), acc);
                float bias = b1[n0 + ln];
#pragma unroll
                for (int r = 0; r < 8; ++r) {
                    int m = r + 8 * hl;
                    shid[m * HSTR + n0 + ln] = (__bf16)eluf(acc[r] + bias);
                }
            }
        }
        __syncthreads();

        // ---- P2: GRU. 6 accumulators per 16-col deter tile, fused gates
        v8f dn[4];
#pragma unroll
        for (int i = 0; i < 4; ++i) {
            int c0 = (wave * 4 + i) * 16;
            v8f aIR = {}, aIZ = {}, aIN = {}, aHR = {}, aHZ = {}, aHN = {};
            for (int kc = 0; kc < 16; ++kc) {
                int kb = kc * 32;
                v16bf ah = load_a_bf16(shid + kb, HSTR, ln, hl);   // h1
                v16bf ad = load_a_f32 (sdet + kb, DSTR, ln, hl);   // old deter
                const WT* wi = Wih + (size_t)kb * (3 * DETER) + c0;
                const WT* wh = Whh + (size_t)kb * (3 * DETER) + c0;
                v16bf bb;
                bb = load_b(wi,             3 * DETER, lane); aIR = wmma_bf(ah, bb, aIR);
                bb = load_b(wi + DETER,     3 * DETER, lane); aIZ = wmma_bf(ah, bb, aIZ);
                bb = load_b(wi + 2 * DETER, 3 * DETER, lane); aIN = wmma_bf(ah, bb, aIN);
                bb = load_b(wh,             3 * DETER, lane); aHR = wmma_bf(ad, bb, aHR);
                bb = load_b(wh + DETER,     3 * DETER, lane); aHZ = wmma_bf(ad, bb, aHZ);
                bb = load_b(wh + 2 * DETER, 3 * DETER, lane); aHN = wmma_bf(ad, bb, aHN);
            }
            int c = c0 + ln;
            float bir = bih[c], biz = bih[DETER + c], bin_ = bih[2 * DETER + c];
            float bhr = bhh[c], bhz = bhh[DETER + c], bhn  = bhh[2 * DETER + c];
#pragma unroll
            for (int r = 0; r < 8; ++r) {
                int m = r + 8 * hl;
                float rg = sigm(aIR[r] + bir + aHR[r] + bhr);
                float zg = sigm(aIZ[r] + biz + aHZ[r] + bhz);
                float nn = tanh_f(aIN[r] + bin_ + rg * (aHN[r] + bhn));
                float h  = sdet[m * DSTR + c];                 // old deter (pre-barrier)
                dn[i][r] = (1.0f - zg) * nn + zg * h;
            }
        }
        __syncthreads();   // all reads of old deter complete
#pragma unroll
        for (int i = 0; i < 4; ++i) {
            int c = (wave * 4 + i) * 16 + ln;
#pragma unroll
            for (int r = 0; r < 8; ++r) {
                int m = r + 8 * hl;
                float v = dn[i][r];
                sdet[m * DSTR + c] = v;                         // deter <- deter_new
                outDT[((size_t)(b0 + m) * T_TOT + t) * DETER + c] = v;
            }
        }
        __syncthreads();

        // ---- P3: h2 = elu(deter_new @ W_img2 + b_img2), 2 tiles/pass ---
#pragma unroll
        for (int ii = 0; ii < 2; ++ii) {
            int n0 = (wave * 4 + ii * 2) * 16;
            int n1 = n0 + 16;
            v8f ac0 = {}, ac1 = {};
            for (int kc = 0; kc < 16; ++kc) {
                int kb = kc * 32;
                v16bf a = load_a_f32(sdet + kb, DSTR, ln, hl);
                ac0 = wmma_bf(a, load_b(W2 + (size_t)kb * HIDDEN + n0, HIDDEN, lane), ac0);
                ac1 = wmma_bf(a, load_b(W2 + (size_t)kb * HIDDEN + n1, HIDDEN, lane), ac1);
            }
            float bi0 = b2[n0 + ln], bi1 = b2[n1 + ln];
#pragma unroll
            for (int r = 0; r < 8; ++r) {
                int m = r + 8 * hl;
                shid[m * HSTR + n0 + ln] = (__bf16)eluf(ac0[r] + bi0);
                shid[m * HSTR + n1 + ln] = (__bf16)eluf(ac1[r] + bi1);
            }
        }
        __syncthreads();

        // ---- P4: prior head (img3): waves 0,1; mean tile + std tile ----
        if (wave < 2) {
            int g = wave;
            v8f am = {}, as = {};
            for (int kc = 0; kc < 16; ++kc) {
                int kb = kc * 32;
                v16bf a = load_a_bf16(shid + kb, HSTR, ln, hl);
                am = wmma_bf(a, load_b(W3 + (size_t)kb * (2 * STOCH) + g * 16,
                                       2 * STOCH, lane), am);
                as = wmma_bf(a, load_b(W3 + (size_t)kb * (2 * STOCH) + STOCH + g * 16,
                                       2 * STOCH, lane), as);
            }
            int c = g * 16 + ln;
            float bm_ = b3[c], bs_ = b3[STOCH + c];
#pragma unroll
            for (int r = 0; r < 8; ++r) {
                int m = r + 8 * hl;
                float mean = am[r] + bm_;
                float sd   = softpf(as[r] + bs_) + 0.1f;
                float eps  = npr[((size_t)t * B_TOT + b0 + m) * STOCH + c];
                float st   = mean + sd * eps;
                size_t oi  = ((size_t)(b0 + m) * T_TOT + t) * STOCH + c;
                outPM[oi] = mean;
                outPS[oi] = sd;
                outPX[oi] = st;
                sxin[m * XSTR + c] = (__bf16)st;   // stoch carry for next step
            }
        }
        __syncthreads();

        // ---- P5: h3 = elu([deter_new|embed] @ W_obs1 + b_obs1), paired -
        {
            const float* et = embed + ((size_t)b0 * T_TOT + t) * EMBED;
#pragma unroll
            for (int ii = 0; ii < 2; ++ii) {
                int n0 = (wave * 4 + ii * 2) * 16;
                int n1 = n0 + 16;
                v8f ac0 = {}, ac1 = {};
                for (int kc = 0; kc < 16; ++kc) {          // deter part (K 0..511)
                    int kb = kc * 32;
                    v16bf a = load_a_f32(sdet + kb, DSTR, ln, hl);
                    ac0 = wmma_bf(a, load_b(Wo1 + (size_t)kb * HIDDEN + n0, HIDDEN, lane), ac0);
                    ac1 = wmma_bf(a, load_b(Wo1 + (size_t)kb * HIDDEN + n1, HIDDEN, lane), ac1);
                }
                for (int kc = 0; kc < 32; ++kc) {          // embed part (K 512..1535)
                    int kb = kc * 32;
                    v16bf a = load_a_f32(et + kb, T_TOT * EMBED, ln, hl);
                    const WT* w = Wo1 + (size_t)(DETER + kb) * HIDDEN;
                    ac0 = wmma_bf(a, load_b(w + n0, HIDDEN, lane), ac0);
                    ac1 = wmma_bf(a, load_b(w + n1, HIDDEN, lane), ac1);
                }
                float bi0 = bo1[n0 + ln], bi1 = bo1[n1 + ln];
#pragma unroll
                for (int r = 0; r < 8; ++r) {
                    int m = r + 8 * hl;
                    shid[m * HSTR + n0 + ln] = (__bf16)eluf(ac0[r] + bi0);
                    shid[m * HSTR + n1 + ln] = (__bf16)eluf(ac1[r] + bi1);
                }
            }
        }
        __syncthreads();

        // ---- P6: posterior head (obs2): waves 0,1 ----------------------
        if (wave < 2) {
            int g = wave;
            v8f am = {}, as = {};
            for (int kc = 0; kc < 16; ++kc) {
                int kb = kc * 32;
                v16bf a = load_a_bf16(shid + kb, HSTR, ln, hl);
                am = wmma_bf(a, load_b(Wo2 + (size_t)kb * (2 * STOCH) + g * 16,
                                       2 * STOCH, lane), am);
                as = wmma_bf(a, load_b(Wo2 + (size_t)kb * (2 * STOCH) + STOCH + g * 16,
                                       2 * STOCH, lane), as);
            }
            int c = g * 16 + ln;
            float bm_ = bo2[c], bs_ = bo2[STOCH + c];
#pragma unroll
            for (int r = 0; r < 8; ++r) {
                int m = r + 8 * hl;
                float mean = am[r] + bm_;
                float sd   = softpf(as[r] + bs_) + 0.1f;
                float eps  = npo[((size_t)t * B_TOT + b0 + m) * STOCH + c];
                float st   = mean + sd * eps;
                size_t oi  = ((size_t)(b0 + m) * T_TOT + t) * STOCH + c;
                outOM[oi] = mean;
                outOS[oi] = sd;
                outOX[oi] = st;
            }
        }
        __syncthreads();
    }
}

extern "C" void kernel_launch(void* const* d_in, const int* in_sizes, int n_in,
                              void* d_out, int out_size, void* d_ws, size_t ws_size,
                              hipStream_t stream) {
    (void)in_sizes; (void)n_in; (void)out_size;
    // setup_inputs order: embed, action, params{W_img1,b_img1,W_ih,b_ih,W_hh,b_hh,
    //   W_img2,b_img2,W_img3,b_img3,W_obs1,b_obs1,W_obs2,b_obs2}, noise_prior, noise_post
    const float* embed  = (const float*)d_in[0];
    const float* action = (const float*)d_in[1];
    const float* W1  = (const float*)d_in[2];
    const float* b1  = (const float*)d_in[3];
    const float* Wih = (const float*)d_in[4];
    const float* bih = (const float*)d_in[5];
    const float* Whh = (const float*)d_in[6];
    const float* bhh = (const float*)d_in[7];
    const float* W2  = (const float*)d_in[8];
    const float* b2  = (const float*)d_in[9];
    const float* W3  = (const float*)d_in[10];
    const float* b3  = (const float*)d_in[11];
    const float* Wo1 = (const float*)d_in[12];
    const float* bo1 = (const float*)d_in[13];
    const float* Wo2 = (const float*)d_in[14];
    const float* bo2 = (const float*)d_in[15];
    const float* npr = (const float*)d_in[16];
    const float* npo = (const float*)d_in[17];
    float* out = (float*)d_out;

    if (ws_size >= WTOT * sizeof(__bf16)) {
        // one-time (per launch) f32 -> bf16 weight conversion into workspace
        __bf16* ws = (__bf16*)d_ws;
        prep_weights_kernel<<<dim3(512), dim3(256), 0, stream>>>(
            W1, Wih, Whh, W2, W3, Wo1, Wo2, ws);
        const __bf16* pW1  = ws;
        const __bf16* pWih = pW1  + S1P;
        const __bf16* pWhh = pWih + SIH;
        const __bf16* pW2  = pWhh + SHH;
        const __bf16* pW3  = pW2  + S2;
        const __bf16* pWo1 = pW3  + S3;
        const __bf16* pWo2 = pWo1 + SO1;
        rssm_rollout_kernel<__bf16><<<dim3(B_TOT / 16), dim3(256), 0, stream>>>(
            embed, action, pW1, b1, pWih, bih, pWhh, bhh, pW2, b2, pW3, b3,
            pWo1, bo1, pWo2, bo2, npr, npo, out);
    } else {
        // fallback: convert weights on the fly from f32
        rssm_rollout_kernel<float><<<dim3(B_TOT / 16), dim3(256), 0, stream>>>(
            embed, action, W1, b1, Wih, bih, Whh, bhh, W2, b2, W3, b3,
            Wo1, bo1, Wo2, bo2, npr, npo, out);
    }
}
